// GroupQueryAttention_79199196938376
// MI455X (gfx1250) — matrix operand order repaired and checked by reading, not compiled, not run on hardware
//
#include <hip/hip_runtime.h>

typedef __bf16 bf16;
typedef __attribute__((ext_vector_type(16))) __bf16 v16bf;
typedef __attribute__((ext_vector_type(8)))  __bf16 v8bf;
typedef __attribute__((ext_vector_type(8)))  float  v8f;
typedef __attribute__((ext_vector_type(4)))  int    v4i;

#define BATCH   2
#define SEQ     2048
#define EMBED   2048
#define NHEAD   16
#define NGROUP  4
#define HDIM    128
#define EKV     512          // NGROUP * HDIM
#define MROWS   (BATCH*SEQ)  // 4096

// ---------------------------------------------------------------------------
// Async global->LDS staging (CDNA5 GLOBAL_LOAD_ASYNC_TO_LDS_B128, ASYNCcnt),
// with a plain-copy fallback if the builtin is unavailable on this toolchain.
// Builtin signature (from hipcc diagnostic): (v4i AS1*, v4i AS3*, imm, imm).
// ---------------------------------------------------------------------------
#if defined(__has_builtin)
#  if __has_builtin(__builtin_amdgcn_global_load_async_to_lds_b128)
#    define ASYNC_LDS_OK 1
#  endif
#endif
#ifndef ASYNC_LDS_OK
#  define ASYNC_LDS_OK 0
#endif

#define ASG __attribute__((address_space(1)))
#define ASL __attribute__((address_space(3)))

__device__ inline void async_copy16(const bf16* g, bf16* l) {
#if ASYNC_LDS_OK
    __builtin_amdgcn_global_load_async_to_lds_b128((ASG v4i*)g, (ASL v4i*)l, 0, 0);
#else
    *(uint4*)l = *(const uint4*)g;
#endif
}

__device__ inline void async_wait() {
#if ASYNC_LDS_OK
#  if __has_builtin(__builtin_amdgcn_s_wait_asynccnt)
    __builtin_amdgcn_s_wait_asynccnt(0);
#  else
    asm volatile("s_wait_asynccnt 0" ::: "memory");
#  endif
#endif
}

// ---------------------------------------------------------------------------
// WMMA helpers (gfx1250, wave32). Layouts per CDNA5 ISA §7.12.2.
// ---------------------------------------------------------------------------
__device__ inline v8f wmma_bf16(v16bf a, v16bf b, v8f c) {
    // (neg_a, A, neg_b, B, c_mod, C, reuse_a, reuse_b)
    return __builtin_amdgcn_wmma_f32_16x16x32_bf16(false, a, false, b,
                                                   (short)0, c, false, false);
}

union V16U { v16bf v; v8bf h[2]; };

// A-matrix 16x32 fragment from row-major tile (stride ld elems):
// lane m=lane&15 holds row m; f[0..7]=A[m][8*hf+j], f[8..15]=A[m][16+8*hf+j]
__device__ inline v16bf ld_a_frag(const bf16* __restrict__ tile, int ld, int lane) {
    int m = lane & 15, hf = lane >> 4;
    const bf16* p = tile + (size_t)m * ld + 8 * hf;
    V16U u;
    u.h[0] = *(const v8bf*)(p);
    u.h[1] = *(const v8bf*)(p + 16);
    return u.v;
}

// B-matrix 32x16 fragment where data is laid out transposed (row = n, K along
// the row, stride ld): f[0..7] = B[8*hf + j][n], f[8..15] = B[16 + 8*hf + j][n].
__device__ inline v16bf ld_b_frag_t(const bf16* __restrict__ tile, int ld, int lane) {
    int n = lane & 15, hf = lane >> 4;
    const bf16* p = tile + (size_t)n * ld + 8 * hf;
    V16U u;
    u.h[0] = *(const v8bf*)(p);
    u.h[1] = *(const v8bf*)(p + 16);
    return u.v;
}

// ---------------------------------------------------------------------------
// fp32 -> bf16 conversion (4 elems/thread; sizes multiple of 1024)
// ---------------------------------------------------------------------------
__global__ __launch_bounds__(256) void f32_to_bf16_kernel(
    const float* __restrict__ src, bf16* __restrict__ dst, int n) {
    int i4 = (blockIdx.x * 256 + threadIdx.x) * 4;
    if (i4 < n) {
        float4 v = *(const float4*)(src + i4);
        dst[i4 + 0] = (bf16)v.x;
        dst[i4 + 1] = (bf16)v.y;
        dst[i4 + 2] = (bf16)v.z;
        dst[i4 + 3] = (bf16)v.w;
    }
}

// ---------------------------------------------------------------------------
// fp32 [K,N] -> bf16 transposed [N,K] (32x32 LDS tile transpose)
// grid: (N/32, K/32), block 256 (= 32 x 8)
// ---------------------------------------------------------------------------
__global__ __launch_bounds__(256) void transpose_f32_to_bf16_kernel(
    const float* __restrict__ src, bf16* __restrict__ dst, int K, int N) {
    __shared__ float t[32][33];
    const int tx = threadIdx.x & 31, ty = threadIdx.x >> 5;
    const int k0 = blockIdx.y * 32, n0 = blockIdx.x * 32;
#pragma unroll
    for (int i = 0; i < 4; ++i) {
        int k = ty + i * 8;
        t[k][tx] = src[(size_t)(k0 + k) * N + n0 + tx];
    }
    __syncthreads();
#pragma unroll
    for (int i = 0; i < 4; ++i) {
        int n = ty + i * 8;
        dst[(size_t)(n0 + n) * K + k0 + tx] = (bf16)t[tx][n];
    }
}

// ---------------------------------------------------------------------------
// Tiled WMMA GEMM:  C[M,N] = (A[M,K](bf16) @ Bt[N,K]^T(bf16) + bias(f32)) * oscale
// Block: 256 threads (8 waves), tile 128x128, BK=32, double-buffered LDS with
// async global->LDS staging. Waves 4(M) x 2(N), each 32x64 (2x4 accum tiles).
// OUTMODE: 0 = bf16 [M,N], 1 = f32 [M,N], 2 = bf16 transposed [N,M].
// ---------------------------------------------------------------------------
template <int OUTMODE>
__global__ __launch_bounds__(256) void gemm_bias_kernel(
    const bf16* __restrict__ A, const bf16* __restrict__ Bt,
    const float* __restrict__ bias, void* __restrict__ C,
    int M, int N, int K, float oscale) {
    __shared__ bf16 As[2][128 * 40];   // 128 rows(m) x 32 k, stride 40
    __shared__ bf16 Bs[2][128 * 40];   // 128 rows(n) x 32 k, stride 40

    const int tid  = threadIdx.x;
    const int wave = tid >> 5, lane = tid & 31;
    const int wm = wave >> 1, wn = wave & 1;
    const int bm = blockIdx.y * 128, bn = blockIdx.x * 128;

    v8f acc[2][4];
#pragma unroll
    for (int i = 0; i < 2; ++i)
#pragma unroll
        for (int j = 0; j < 4; ++j)
            acc[i][j] = (v8f){0.f, 0.f, 0.f, 0.f, 0.f, 0.f, 0.f, 0.f};

    auto load_tiles = [&](int k0, int buf) {
#pragma unroll
        for (int i = 0; i < 2; ++i) {
            int c = tid + i * 256;             // 512 chunks of 8 bf16 per tile
            int r = c >> 2, cl = (c & 3) * 8;  // tile 128 x 32
            async_copy16(&A [(size_t)(bm + r) * K + k0 + cl], &As[buf][r * 40 + cl]);
            async_copy16(&Bt[(size_t)(bn + r) * K + k0 + cl], &Bs[buf][r * 40 + cl]);
        }
    };

    const int NK = K / 32;
    load_tiles(0, 0);
    async_wait();
    __syncthreads();

    for (int kt = 0; kt < NK; ++kt) {
        const int buf = kt & 1;
        if (kt + 1 < NK) load_tiles((kt + 1) * 32, buf ^ 1);

        v16bf af[2], bf_[4];
#pragma unroll
        for (int i = 0; i < 2; ++i)
            af[i] = ld_a_frag(&As[buf][(size_t)(wm * 32 + i * 16) * 40], 40, lane);
#pragma unroll
        for (int j = 0; j < 4; ++j)
            bf_[j] = ld_b_frag_t(&Bs[buf][(size_t)(wn * 64 + j * 16) * 40], 40, lane);
#pragma unroll
        for (int i = 0; i < 2; ++i)
#pragma unroll
            for (int j = 0; j < 4; ++j)
                acc[i][j] = wmma_bf16(af[i], bf_[j], acc[i][j]);

        if (kt + 1 < NK) {
            async_wait();
            __syncthreads();
        }
    }

    const int ln = lane & 15, hf = lane >> 4;
#pragma unroll
    for (int i = 0; i < 2; ++i)
#pragma unroll
        for (int j = 0; j < 4; ++j) {
            int gn = bn + wn * 64 + j * 16 + ln;
            float bv = bias[gn];
#pragma unroll
            for (int r = 0; r < 8; ++r) {
                int gm = bm + wm * 32 + i * 16 + r + 8 * hf;
                float val = (acc[i][j][r] + bv) * oscale;
                if (OUTMODE == 1)
                    ((float*)C)[(size_t)gm * N + gn] = val;
                else if (OUTMODE == 2)
                    ((bf16*)C)[(size_t)gn * M + gm] = (bf16)val;   // transposed
                else
                    ((bf16*)C)[(size_t)gm * N + gn] = (bf16)val;
            }
        }
}

// ---------------------------------------------------------------------------
// GQA flash attention.
// Grid: (SEQ/128, BATCH*NHEAD). Block: 256 threads = 8 waves, 16 q-rows/wave.
// Q: [B*S, EMBED] bf16, PRE-SCALED by 1/sqrt(HDIM) at projection time.
// K: [B*S, EKV] bf16 (col = g*128+d).  Vt: [EKV, B*S] bf16 (transposed).
// O: [B*S, EMBED] bf16.
// No occupancy cap: large accumulator state intentionally uses CDNA5's
// >256-VGPR mode (s_set_vgpr_msb) instead of spilling to scratch.
// ---------------------------------------------------------------------------
__global__ __launch_bounds__(256) void gqa_flash_kernel(
    const bf16* __restrict__ Q, const bf16* __restrict__ K,
    const bf16* __restrict__ Vt, bf16* __restrict__ O) {
    __shared__ bf16 Ks[64 * 136];    // 64 kv rows x 128 hd
    __shared__ bf16 Vs[128 * 72];    // 128 hd rows x 64 kv (transposed V)
    __shared__ bf16 Ps[8 * 16 * 64]; // per-wave 16x64 P tile

    const int tid = threadIdx.x, wave = tid >> 5, lane = tid & 31;
    const int ln = lane & 15, hf = lane >> 4;
    const int qt = blockIdx.x, bh = blockIdx.y;
    const int b = bh >> 4, h = bh & 15, g = h >> 2;
    const size_t qrow0 = (size_t)b * SEQ + qt * 128 + wave * 16;

    // Resident Q fragments: 4 k-chunks of 32 over head_dim=128
    v16bf qf[4];
    {
        const bf16* qb = Q + (qrow0 + ln) * EMBED + h * HDIM;
#pragma unroll
        for (int kk = 0; kk < 4; ++kk) {
            const bf16* p = qb + kk * 32 + 8 * hf;
            V16U u;
            u.h[0] = *(const v8bf*)(p);
            u.h[1] = *(const v8bf*)(p + 16);
            qf[kk] = u.v;
        }
    }

    v8f acc[8];
#pragma unroll
    for (int j = 0; j < 8; ++j)
        acc[j] = (v8f){0.f, 0.f, 0.f, 0.f, 0.f, 0.f, 0.f, 0.f};
    float mi[8], li[8];
#pragma unroll
    for (int r = 0; r < 8; ++r) { mi[r] = -1e30f; li[r] = 0.f; }

    bf16* myP = Ps + wave * 16 * 64;
    const int NKT = SEQ / 64;

    for (int kt = 0; kt < NKT; ++kt) {
        __syncthreads();   // all waves done reading previous tile
        const bf16* kg = K  + ((size_t)b * SEQ + kt * 64) * EKV + g * HDIM;
        const bf16* vg = Vt + (size_t)g * HDIM * MROWS + (size_t)b * SEQ + kt * 64;
#pragma unroll
        for (int i = 0; i < 4; ++i) {
            int c = tid + i * 256;
            {   // K tile: 64 rows(kv) x 128 cols(hd), chunks of 8
                int row = c >> 4, col = (c & 15) * 8;
                async_copy16(&kg[(size_t)row * EKV + col], &Ks[row * 136 + col]);
                if (kt + 1 < NKT)
                    __builtin_prefetch(&kg[(size_t)(row + 64) * EKV + col], 0, 1);
            }
            {   // Vt tile: 128 rows(hd) x 64 cols(kv), chunks of 8
                int row = c >> 3, col = (c & 7) * 8;
                async_copy16(&vg[(size_t)row * MROWS + col], &Vs[row * 72 + col]);
                if (kt + 1 < NKT)
                    __builtin_prefetch(&vg[(size_t)row * MROWS + col + 64], 0, 1);
            }
        }
        async_wait();
        __syncthreads();

        // scores S[16 x 64] = Q @ K^T (4 n-tiles x 4 k-steps);
        // B = K^T fragment: f[0..7] = Ks[nt*16 + n][kk*32 + 8*hf + j], f[8..15] at +16
        v8f s[4];
#pragma unroll
        for (int nt = 0; nt < 4; ++nt) {
            v8f sa = (v8f){0.f, 0.f, 0.f, 0.f, 0.f, 0.f, 0.f, 0.f};
#pragma unroll
            for (int kk = 0; kk < 4; ++kk) {
                const bf16* p = Ks + (size_t)(nt * 16 + ln) * 136 + kk * 32 + 8 * hf;
                V16U u;
                u.h[0] = *(const v8bf*)(p);
                u.h[1] = *(const v8bf*)(p + 16);
                sa = wmma_bf16(qf[kk], u.v, sa);
            }
            s[nt] = sa;
        }

        // online softmax: row m = r + 8*hf lives across 16 lanes (same r,hf).
        // Q was pre-scaled by 1/sqrt(HDIM), so s is already the scaled score.
        // acc rescale is folded into this loop (no alpha[] array -> less regs).
#pragma unroll
        for (int r = 0; r < 8; ++r) {
            float f0 = s[0][r], f1 = s[1][r];
            float f2 = s[2][r], f3 = s[3][r];
            float mx = fmaxf(fmaxf(f0, f1), fmaxf(f2, f3));
#pragma unroll
            for (int o = 1; o < 16; o <<= 1)
                mx = fmaxf(mx, __shfl_xor(mx, o, 32));
            float mn = fmaxf(mi[r], mx);
            float al = __expf(mi[r] - mn);
            float p0 = __expf(f0 - mn), p1 = __expf(f1 - mn);
            float p2 = __expf(f2 - mn), p3 = __expf(f3 - mn);
            bf16* pw = myP + (size_t)(r + 8 * hf) * 64 + ln;
            pw[0]  = (bf16)p0;
            pw[16] = (bf16)p1;
            pw[32] = (bf16)p2;
            pw[48] = (bf16)p3;
            float rs = p0 + p1 + p2 + p3;
#pragma unroll
            for (int o = 1; o < 16; o <<= 1)
                rs += __shfl_xor(rs, o, 32);
            li[r] = li[r] * al + rs;
            mi[r] = mn;
#pragma unroll
            for (int j = 0; j < 8; ++j)
                acc[j][r] *= al;
        }

        // out += P[16x64] @ V[64x128]  (2 k-steps x 8 n-tiles);
        // V B-fragment from transposed Vs: f[0..7] = Vs[j16+n][kk*32 + 8*hf + j], f[8..15] at +16
#pragma unroll
        for (int kk = 0; kk < 2; ++kk) {
            const bf16* pr = myP + (size_t)ln * 64 + kk * 32 + 8 * hf;
            V16U u;
            u.h[0] = *(const v8bf*)(pr);
            u.h[1] = *(const v8bf*)(pr + 16);
            v16bf af = u.v;
#pragma unroll
            for (int j = 0; j < 8; ++j) {
                const bf16* p = Vs + (size_t)(j * 16 + ln) * 72 + kk * 32 + 8 * hf;
                V16U w;
                w.h[0] = *(const v8bf*)(p);
                w.h[1] = *(const v8bf*)(p + 16);
                acc[j] = wmma_bf16(af, w.v, acc[j]);
            }
        }
    }

    // epilogue: normalize by li, store bf16 at [row, h*128 + d]
#pragma unroll
    for (int j = 0; j < 8; ++j) {
#pragma unroll
        for (int r = 0; r < 8; ++r) {
            size_t row = qrow0 + r + 8 * hf;
            float v = acc[j][r] / li[r];
            O[row * EMBED + h * HDIM + j * 16 + ln] = (bf16)v;
        }
    }
}

// ---------------------------------------------------------------------------
// Host-side launcher
// ---------------------------------------------------------------------------
extern "C" void kernel_launch(void* const* d_in, const int* in_sizes, int n_in,
                              void* d_out, int out_size, void* d_ws, size_t ws_size,
                              hipStream_t stream) {
    const float* x  = (const float*)d_in[0];
    const float* Wq = (const float*)d_in[1];
    const float* bq = (const float*)d_in[2];
    const float* Wk = (const float*)d_in[3];
    const float* bk = (const float*)d_in[4];
    const float* Wv = (const float*)d_in[5];
    const float* bv = (const float*)d_in[6];
    const float* Wo = (const float*)d_in[7];
    const float* bo = (const float*)d_in[8];
    float* out = (float*)d_out;

    char* ws = (char*)d_ws;
    size_t off = 0;
    auto carve = [&](size_t elems) {
        bf16* p = (bf16*)(ws + off);
        off += ((elems * 2 + 255) & ~(size_t)255);
        return p;
    };
    bf16* xb  = carve((size_t)MROWS * EMBED);
    bf16* Wqt = carve((size_t)EMBED * EMBED);   // [N=EMBED, K=EMBED]
    bf16* Wkt = carve((size_t)EKV * EMBED);     // [N=EKV,   K=EMBED]
    bf16* Wvt = carve((size_t)EKV * EMBED);
    bf16* Wot = carve((size_t)EMBED * EMBED);
    bf16* Qb  = carve((size_t)MROWS * EMBED);
    bf16* Kb  = carve((size_t)MROWS * EKV);
    bf16* Vtb = carve((size_t)EKV * MROWS);     // transposed V
    bf16* Ab  = carve((size_t)MROWS * EMBED);

    {   // x -> bf16
        size_t n = (size_t)MROWS * EMBED;
        f32_to_bf16_kernel<<<(int)((n / 4 + 255) / 256), 256, 0, stream>>>(x, xb, (int)n);
    }
    // weights -> bf16 transposed [N,K]
    transpose_f32_to_bf16_kernel<<<dim3(EMBED / 32, EMBED / 32), 256, 0, stream>>>(Wq, Wqt, EMBED, EMBED);
    transpose_f32_to_bf16_kernel<<<dim3(EKV / 32,   EMBED / 32), 256, 0, stream>>>(Wk, Wkt, EMBED, EKV);
    transpose_f32_to_bf16_kernel<<<dim3(EKV / 32,   EMBED / 32), 256, 0, stream>>>(Wv, Wvt, EMBED, EKV);
    transpose_f32_to_bf16_kernel<<<dim3(EMBED / 32, EMBED / 32), 256, 0, stream>>>(Wo, Wot, EMBED, EMBED);

    const float qscale = 0.08838834764831845f;  // 1/sqrt(HDIM), folded into Q
    // projections
    gemm_bias_kernel<0><<<dim3(EMBED / 128, MROWS / 128), 256, 0, stream>>>(
        xb, Wqt, bq, Qb, MROWS, EMBED, EMBED, qscale);
    gemm_bias_kernel<0><<<dim3(EKV / 128, MROWS / 128), 256, 0, stream>>>(
        xb, Wkt, bk, Kb, MROWS, EKV, EMBED, 1.0f);
    gemm_bias_kernel<2><<<dim3(EKV / 128, MROWS / 128), 256, 0, stream>>>(
        xb, Wvt, bv, Vtb, MROWS, EKV, EMBED, 1.0f);   // V stored transposed [EKV, MROWS]

    // attention
    gqa_flash_kernel<<<dim3(SEQ / 128, BATCH * NHEAD), 256, 0, stream>>>(
        Qb, Kb, Vtb, Ab);

    // output projection (fp32 out)
    gemm_bias_kernel<1><<<dim3(EMBED / 128, MROWS / 128), 256, 0, stream>>>(
        Ab, Wot, bo, out, MROWS, EMBED, EMBED, 1.0f);
}
